// DisentangledSelfAttention_17093969838352
// MI455X (gfx1250) — matrix operand order repaired and checked by reading, not compiled, not run on hardware
//
#include <hip/hip_runtime.h>

// Problem constants (from reference)
#define BB   8
#define SSEQ 512
#define HIDN 1024
#define NHD  16
#define DHEAD 64
// scale = 1/sqrt(64*3)
#define SCALE 0.07216878364870323f

typedef __attribute__((ext_vector_type(16))) _Float16 v16h;
typedef __attribute__((ext_vector_type(8)))  float    v8f;

__device__ __forceinline__ v8f wmma16(v16h a, v16h b, v8f c) {
  // v_wmma_f32_16x16x32_f16 : D = A(16x32 f16) * B(32x16 f16) + C(16x16 f32)
  return __builtin_amdgcn_wmma_f32_16x16x32_f16(false, a, false, b, (short)0, c,
                                                false, false);
}

// A fragment (16x32 f16). Lane: m = lane%16, hf = lane/16.
// VGPR0-3 hold K = kbase + hf*8 + [0..7]; VGPR4-7 hold +16.
__device__ __forceinline__ v16h ldA_f16(const _Float16* __restrict__ row,
                                        int kbase, int hf) {
  v16h a;
  int k1 = kbase + hf * 8;
  int k2 = kbase + 16 + hf * 8;
#pragma unroll
  for (int i = 0; i < 8; i++) {
    a[i]     = row[k1 + i];
    a[8 + i] = row[k2 + i];
  }
  return a;
}

// B fragment (32x16 f16): lane column n = lane%16, K = kbase + hf*8 + [0..7]
// then kbase + 16 + hf*8 + [0..7] from this lane's row pointer (K contiguous).
__device__ __forceinline__ v16h ldB_f16(const _Float16* __restrict__ row,
                                        int kbase, int hf) {
  v16h b;
  int k1 = kbase + hf * 8;
  int k2 = kbase + 16 + hf * 8;
#pragma unroll
  for (int i = 0; i < 8; i++) {
    b[i]     = row[k1 + i];
    b[8 + i] = row[k2 + i];
  }
  return b;
}

// ---------------------------------------------------------------------------
// Elementwise f32 -> f16 convert (one-time operand prepass).
// ---------------------------------------------------------------------------
__global__ void k_cvt(const float* __restrict__ src, _Float16* __restrict__ dst,
                      int n) {
  int i = blockIdx.x * blockDim.x + threadIdx.x;
  if (i < n) dst[i] = (_Float16)src[i];
}

// ---------------------------------------------------------------------------
// YH[M,N] f16 = ( XH[M,K] @ WH[N,K]^T + bias ) * outscale  (torch Linear)
// All-f16 operands, f32 accumulate. One wave: 32(m) x 64(n) tile
// (2 A frags x 4 B frags -> 8 independent WMMAs per k-step).
// ---------------------------------------------------------------------------
__global__ __launch_bounds__(128) void k_gemm_f16(
    const _Float16* __restrict__ X, const _Float16* __restrict__ W,
    const float* __restrict__ bias, _Float16* __restrict__ Y,
    int M, int N, int K, float outscale) {
  int lane = threadIdx.x & 31;
  int wave = blockIdx.x * (blockDim.x >> 5) + (threadIdx.x >> 5);
  int tiles_n = N >> 6;
  int mt = wave / tiles_n;
  int nt = wave - mt * tiles_n;
  int row0 = mt << 5, col0 = nt << 6;
  if (row0 >= M) return;  // wave-uniform
  int hf = lane >> 4, nl = lane & 15;

  v8f c00 = {}, c01 = {}, c02 = {}, c03 = {};
  v8f c10 = {}, c11 = {}, c12 = {}, c13 = {};
  const _Float16* x0 = X + (size_t)(row0 + nl) * K;
  const _Float16* x1 = X + (size_t)(row0 + 16 + nl) * K;
  const _Float16* w0 = W + (size_t)(col0 + nl) * K;
  const _Float16* w1 = W + (size_t)(col0 + 16 + nl) * K;
  const _Float16* w2 = W + (size_t)(col0 + 32 + nl) * K;
  const _Float16* w3 = W + (size_t)(col0 + 48 + nl) * K;

  for (int k = 0; k < K; k += 32) {
    v16h a0 = ldA_f16(x0, k, hf);
    v16h a1 = ldA_f16(x1, k, hf);
    v16h b0 = ldB_f16(w0, k, hf);
    v16h b1 = ldB_f16(w1, k, hf);
    v16h b2 = ldB_f16(w2, k, hf);
    v16h b3 = ldB_f16(w3, k, hf);
    c00 = wmma16(a0, b0, c00);
    c10 = wmma16(a1, b0, c10);
    c01 = wmma16(a0, b1, c01);
    c11 = wmma16(a1, b1, c11);
    c02 = wmma16(a0, b2, c02);
    c12 = wmma16(a1, b2, c12);
    c03 = wmma16(a0, b3, c03);
    c13 = wmma16(a1, b3, c13);
  }
  float bb0 = bias[col0 + nl];
  float bb1 = bias[col0 + 16 + nl];
  float bb2 = bias[col0 + 32 + nl];
  float bb3 = bias[col0 + 48 + nl];
#pragma unroll
  for (int v = 0; v < 8; v++) {
    int m = v + hf * 8;
    _Float16* yr0 = Y + (size_t)(row0 + m) * N + col0;
    _Float16* yr1 = Y + (size_t)(row0 + 16 + m) * N + col0;
    yr0[nl]      = (_Float16)((c00[v] + bb0) * outscale);
    yr0[16 + nl] = (_Float16)((c01[v] + bb1) * outscale);
    yr0[32 + nl] = (_Float16)((c02[v] + bb2) * outscale);
    yr0[48 + nl] = (_Float16)((c03[v] + bb3) * outscale);
    yr1[nl]      = (_Float16)((c10[v] + bb0) * outscale);
    yr1[16 + nl] = (_Float16)((c11[v] + bb1) * outscale);
    yr1[32 + nl] = (_Float16)((c12[v] + bb2) * outscale);
    yr1[48 + nl] = (_Float16)((c13[v] + bb3) * outscale);
  }
}

// ---------------------------------------------------------------------------
// Positional score GEMM (per b,h):
//   OUT[(b*NH+h)*S + q][j] = sum_d A[b*S+q, h*64+d] * P[512+j, h*64+d]
// f16 in/out. Wave: 32(q) x 64(j) tile, K = 64 (2 k-steps x 8 WMMAs).
// ---------------------------------------------------------------------------
__global__ __launch_bounds__(128) void k_posdot(
    const _Float16* __restrict__ A, const _Float16* __restrict__ P,
    _Float16* __restrict__ OUT) {
  int b = blockIdx.z, h = blockIdx.y;
  int lane = threadIdx.x & 31;
  int wave = blockIdx.x * (blockDim.x >> 5) + (threadIdx.x >> 5);  // 0..127
  int mt = wave >> 3;        // 16 q-tiles of 32
  int nt = wave & 7;         // 8 j-tiles of 64
  int q0 = mt << 5, j0 = nt << 6;
  int hf = lane >> 4, nl = lane & 15;

  const _Float16* a0r = A + (size_t)(b * SSEQ + q0 + nl) * HIDN + h * DHEAD;
  const _Float16* a1r = A + (size_t)(b * SSEQ + q0 + 16 + nl) * HIDN + h * DHEAD;
  const _Float16* p0 = P + (size_t)(512 + j0 + nl) * HIDN + h * DHEAD;
  const _Float16* p1 = p0 + (size_t)16 * HIDN;
  const _Float16* p2 = p0 + (size_t)32 * HIDN;
  const _Float16* p3 = p0 + (size_t)48 * HIDN;

  v8f c00 = {}, c01 = {}, c02 = {}, c03 = {};
  v8f c10 = {}, c11 = {}, c12 = {}, c13 = {};
#pragma unroll
  for (int k = 0; k < DHEAD; k += 32) {
    v16h a0 = ldA_f16(a0r, k, hf);
    v16h a1 = ldA_f16(a1r, k, hf);
    v16h b0 = ldB_f16(p0, k, hf);
    v16h b1 = ldB_f16(p1, k, hf);
    v16h b2 = ldB_f16(p2, k, hf);
    v16h b3 = ldB_f16(p3, k, hf);
    c00 = wmma16(a0, b0, c00);
    c10 = wmma16(a1, b0, c10);
    c01 = wmma16(a0, b1, c01);
    c11 = wmma16(a1, b1, c11);
    c02 = wmma16(a0, b2, c02);
    c12 = wmma16(a1, b2, c12);
    c03 = wmma16(a0, b3, c03);
    c13 = wmma16(a1, b3, c13);
  }
  _Float16* obase = OUT + ((size_t)(b * NHD + h) * SSEQ) * 512;
#pragma unroll
  for (int v = 0; v < 8; v++) {
    int m = v + hf * 8;
    _Float16* or0 = obase + (size_t)(q0 + m) * 512 + j0;
    _Float16* or1 = obase + (size_t)(q0 + 16 + m) * 512 + j0;
    or0[nl]      = (_Float16)c00[v];
    or0[16 + nl] = (_Float16)c01[v];
    or0[32 + nl] = (_Float16)c02[v];
    or0[48 + nl] = (_Float16)c03[v];
    or1[nl]      = (_Float16)c10[v];
    or1[16 + nl] = (_Float16)c11[v];
    or1[32 + nl] = (_Float16)c12[v];
    or1[48 + nl] = (_Float16)c13[v];
  }
}

// ---------------------------------------------------------------------------
// V transpose: VT[((b*NH+h)*64 + d)*512 + s] = VH[(b*S+s)*HID + h*64+d]
// ---------------------------------------------------------------------------
__global__ void k_vt(const _Float16* __restrict__ VH, _Float16* __restrict__ VT) {
  size_t i = (size_t)blockIdx.x * blockDim.x + threadIdx.x;
  const size_t total = (size_t)BB * NHD * DHEAD * SSEQ;
  if (i >= total) return;
  int s = (int)(i & 511);
  size_t r = i >> 9;
  int d = (int)(r & 63); r >>= 6;
  int h = (int)(r & 15);
  int b = (int)(r >> 4);
  VT[i] = VH[(size_t)(b * SSEQ + s) * HIDN + h * DHEAD + d];
}

// ---------------------------------------------------------------------------
// Flash attention per (b, h, 16-row q strip). Block = 4 waves = 64 q rows.
// QH is pre-scaled by SCALE (covers content + c2p); PQ output pre-scaled
// (covers p2c) -> score = QK + C2P[q, q-k] + P2C[k, q-k] directly.
// ---------------------------------------------------------------------------
__global__ __launch_bounds__(128) void k_attn(
    const _Float16* __restrict__ QH, const _Float16* __restrict__ KH,
    const _Float16* __restrict__ C2P, const _Float16* __restrict__ P2C,
    const _Float16* __restrict__ VT, float* __restrict__ OUT) {
  __shared__ _Float16 lds_p[4][16 * 32];

  int b = blockIdx.z, h = blockIdx.y;
  int wid = threadIdx.x >> 5;
  int lane = threadIdx.x & 31;
  int hf = lane >> 4, nl = lane & 15;
  int q0 = blockIdx.x * 64 + wid * 16;
  int bh = b * NHD + h;

  const _Float16* qrow = QH + (size_t)(b * SSEQ + q0 + nl) * HIDN + h * DHEAD;
  v16h aq0 = ldA_f16(qrow, 0, hf);
  v16h aq1 = ldA_f16(qrow, 32, hf);

  const _Float16* c2p = C2P + (size_t)bh * SSEQ * 512;
  const _Float16* p2c = P2C + (size_t)bh * SSEQ * 512;
  const _Float16* vt  = VT + (size_t)bh * DHEAD * 512;

  float mrow[8], lrow[8];
#pragma unroll
  for (int v = 0; v < 8; v++) { mrow[v] = -3.0e38f; lrow[v] = 0.0f; }
  v8f o0 = {}, o1 = {}, o2 = {}, o3 = {};

  int nchunk = ((q0 + 15) >> 5) + 1;  // 32-key chunks; k0 <= q0 always
  for (int c = 0; c < nchunk; c++) {
    int k0 = c << 5;
    // ---- content scores: two 16x16 tiles over 32 keys ----
    v8f s0 = {}, s1 = {};
    {
      const _Float16* kr0 = KH + (size_t)(b * SSEQ + k0 + nl) * HIDN + h * DHEAD;
      const _Float16* kr1 = kr0 + (size_t)16 * HIDN;
      s0 = wmma16(aq0, ldB_f16(kr0, 0, hf), s0);
      s0 = wmma16(aq1, ldB_f16(kr0, 32, hf), s0);
      s1 = wmma16(aq0, ldB_f16(kr1, 0, hf), s1);
      s1 = wmma16(aq1, ldB_f16(kr1, 32, hf), s1);
    }
    // ---- add positional terms, mask, online softmax update ----
    float e0s[8], e1s[8];
#pragma unroll
    for (int v = 0; v < 8; v++) {
      int q = q0 + v + hf * 8;
      int key0 = k0 + nl;
      int key1 = k0 + 16 + nl;
      float x0 = -3.0e38f, x1 = -3.0e38f;
      if (key0 <= q) {
        int dlt = q - key0;
        x0 = s0[v] + (float)c2p[(size_t)q * 512 + dlt]
                   + (float)p2c[(size_t)key0 * 512 + dlt];
      }
      if (key1 <= q) {
        int dlt = q - key1;
        x1 = s1[v] + (float)c2p[(size_t)q * 512 + dlt]
                   + (float)p2c[(size_t)key1 * 512 + dlt];
      }
      float tmax = fmaxf(x0, x1);
#pragma unroll
      for (int off = 1; off < 16; off <<= 1)
        tmax = fmaxf(tmax, __shfl_xor(tmax, off, 32));
      float nm = fmaxf(mrow[v], tmax);
      float e0 = __expf(x0 - nm);
      float e1 = __expf(x1 - nm);
      float esum = e0 + e1;
#pragma unroll
      for (int off = 1; off < 16; off <<= 1)
        esum += __shfl_xor(esum, off, 32);
      float corr = __expf(mrow[v] - nm);
      lrow[v] = lrow[v] * corr + esum;
      mrow[v] = nm;
      o0[v] *= corr; o1[v] *= corr; o2[v] *= corr; o3[v] *= corr;
      e0s[v] = e0; e1s[v] = e1;
    }
    // ---- reshape P (C-layout -> A-layout) through per-wave LDS ----
    _Float16* lp = &lds_p[wid][0];
#pragma unroll
    for (int v = 0; v < 8; v++) {
      int m = v + hf * 8;
      lp[m * 32 + nl]      = (_Float16)e0s[v];
      lp[m * 32 + 16 + nl] = (_Float16)e1s[v];
    }
    asm volatile("s_wait_dscnt 0x0" ::: "memory");
    v16h ap;
    {
      const _Float16* pr = lp + nl * 32;
      int k1 = hf * 8;
#pragma unroll
      for (int i = 0; i < 8; i++) {
        ap[i]     = pr[k1 + i];
        ap[8 + i] = pr[16 + k1 + i];
      }
    }
    // ---- O += P @ V  (V pre-transposed f16: rows = d, contiguous keys) ----
    const _Float16* vr0 = vt + (size_t)(0 * 16 + nl) * 512 + k0;
    const _Float16* vr1 = vt + (size_t)(1 * 16 + nl) * 512 + k0;
    const _Float16* vr2 = vt + (size_t)(2 * 16 + nl) * 512 + k0;
    const _Float16* vr3 = vt + (size_t)(3 * 16 + nl) * 512 + k0;
    o0 = wmma16(ap, ldB_f16(vr0, 0, hf), o0);
    o1 = wmma16(ap, ldB_f16(vr1, 0, hf), o1);
    o2 = wmma16(ap, ldB_f16(vr2, 0, hf), o2);
    o3 = wmma16(ap, ldB_f16(vr3, 0, hf), o3);
  }
  // ---- normalize and store [B,S,NH,D] f32 ----
#pragma unroll
  for (int v = 0; v < 8; v++) {
    int m = v + hf * 8;
    float inv = 1.0f / lrow[v];
    float* orow = OUT + (size_t)(b * SSEQ + q0 + m) * HIDN + h * DHEAD;
    orow[nl]      = o0[v] * inv;
    orow[16 + nl] = o1[v] * inv;
    orow[32 + nl] = o2[v] * inv;
    orow[48 + nl] = o3[v] * inv;
  }
}

// ---------------------------------------------------------------------------
extern "C" void kernel_launch(void* const* d_in, const int* in_sizes, int n_in,
                              void* d_out, int out_size, void* d_ws, size_t ws_size,
                              hipStream_t stream) {
  (void)in_sizes; (void)n_in; (void)out_size; (void)ws_size;
  const float* q   = (const float*)d_in[0];
  const float* kx  = (const float*)d_in[1];
  const float* vx  = (const float*)d_in[2];
  // d_in[3] = attention_mask (causal tril) — implicit in the kernels
  const float* Wq  = (const float*)d_in[4];
  const float* bq  = (const float*)d_in[5];
  const float* Wk  = (const float*)d_in[6];
  const float* bk  = (const float*)d_in[7];
  const float* Wv  = (const float*)d_in[8];
  const float* bv  = (const float*)d_in[9];
  const float* Wpk = (const float*)d_in[10];
  const float* bpk = (const float*)d_in[11];
  const float* Wpq = (const float*)d_in[12];
  const float* bpq = (const float*)d_in[13];
  const float* rel = (const float*)d_in[14];

  char* ws = (char*)d_ws;
  const size_t SZ_ACT = (size_t)BB * SSEQ * HIDN * sizeof(_Float16);        // 8 MB
  const size_t SZ_POS = (size_t)2 * 512 * HIDN * sizeof(_Float16);          // 2 MB
  const size_t SZ_SC  = (size_t)BB * NHD * SSEQ * 512 * sizeof(_Float16);   // 64 MB
  size_t off = 0;
  _Float16* QH   = (_Float16*)(ws + off); off += SZ_ACT;
  _Float16* KH   = (_Float16*)(ws + off); off += SZ_ACT;
  _Float16* VH   = (_Float16*)(ws + off); off += SZ_ACT;
  _Float16* PKH  = (_Float16*)(ws + off); off += SZ_POS;
  _Float16* PQH  = (_Float16*)(ws + off); off += SZ_POS;
  _Float16* C2P  = (_Float16*)(ws + off); off += SZ_SC;
  _Float16* P2C  = (_Float16*)(ws + off); off += SZ_SC;
  _Float16* VT   = (_Float16*)(ws + off); off += SZ_ACT;
  _Float16* XQ   = (_Float16*)(ws + off); off += SZ_ACT;
  _Float16* XK   = (_Float16*)(ws + off); off += SZ_ACT;
  _Float16* XV   = (_Float16*)(ws + off); off += SZ_ACT;
  _Float16* RELH = (_Float16*)(ws + off); off += SZ_POS;
  _Float16* WQH  = (_Float16*)(ws + off); off += SZ_POS;
  _Float16* WKH  = (_Float16*)(ws + off); off += SZ_POS;
  _Float16* WVH  = (_Float16*)(ws + off); off += SZ_POS;
  _Float16* WPKH = (_Float16*)(ws + off); off += SZ_POS;
  _Float16* WPQH = (_Float16*)(ws + off); off += SZ_POS;

  const int N_ACT = BB * SSEQ * HIDN;   // 4M
  const int N_POS = 1024 * HIDN;        // 1M

  // ---- one-time f32 -> f16 operand conversion ----
  k_cvt<<<(N_ACT + 255) / 256, 256, 0, stream>>>(q,   XQ,   N_ACT);
  k_cvt<<<(N_ACT + 255) / 256, 256, 0, stream>>>(kx,  XK,   N_ACT);
  k_cvt<<<(N_ACT + 255) / 256, 256, 0, stream>>>(vx,  XV,   N_ACT);
  k_cvt<<<(N_POS + 255) / 256, 256, 0, stream>>>(rel, RELH, N_POS);
  k_cvt<<<(N_POS + 255) / 256, 256, 0, stream>>>(Wq,  WQH,  N_POS);
  k_cvt<<<(N_POS + 255) / 256, 256, 0, stream>>>(Wk,  WKH,  N_POS);
  k_cvt<<<(N_POS + 255) / 256, 256, 0, stream>>>(Wv,  WVH,  N_POS);
  k_cvt<<<(N_POS + 255) / 256, 256, 0, stream>>>(Wpk, WPKH, N_POS);
  k_cvt<<<(N_POS + 255) / 256, 256, 0, stream>>>(Wpq, WPQH, N_POS);

  const int M_BIG = BB * SSEQ;  // 4096
  // waves = (M/32)*(N/64), 4 waves/block
  int blk_big = (M_BIG / 32) * (HIDN / 64) / 4;   // 512
  int blk_pos = (1024 / 32) * (HIDN / 64) / 4;    // 128
  // Projections (QH pre-scaled; PQ output pre-scaled -> p2c term scaled)
  k_gemm_f16<<<blk_big, 128, 0, stream>>>(XQ,   WQH,  bq,  QH,  M_BIG, HIDN, HIDN, SCALE);
  k_gemm_f16<<<blk_big, 128, 0, stream>>>(XK,   WKH,  bk,  KH,  M_BIG, HIDN, HIDN, 1.0f);
  k_gemm_f16<<<blk_big, 128, 0, stream>>>(XV,   WVH,  bv,  VH,  M_BIG, HIDN, HIDN, 1.0f);
  k_gemm_f16<<<blk_pos, 128, 0, stream>>>(RELH, WPKH, bpk, PKH, 1024,  HIDN, HIDN, 1.0f);
  k_gemm_f16<<<blk_pos, 128, 0, stream>>>(RELH, WPQH, bpq, PQH, 1024,  HIDN, HIDN, SCALE);

  // Positional score tables (f16, L2-resident)
  k_posdot<<<dim3(32, NHD, BB), 128, 0, stream>>>(QH, PKH, C2P);
  k_posdot<<<dim3(32, NHD, BB), 128, 0, stream>>>(KH, PQH, P2C);

  // V transpose to f16 [B,NH,D,S]
  k_vt<<<(BB * NHD * DHEAD * SSEQ) / 256, 256, 0, stream>>>(VH, VT);

  // Fused flash attention
  k_attn<<<dim3(SSEQ / 64, NHD, BB), 128, 0, stream>>>(QH, KH, C2P, P2C, VT,
                                                       (float*)d_out);
}
